// MoERouter_24189255811772
// MI455X (gfx1250) — matrix operand order repaired and checked by reading, not compiled, not run on hardware
//
#include <hip/hip_runtime.h>
#include <stdint.h>

#define DIM      4096
#define NEXP     64
#define NTOK     16384
#define KTILES   128          // 4096 / 32
// d_out layout (floats): topk_ids[16384*8] | topk_probs[16384*8] | shared_ids[16384*2] | shared_probs[16384*2] | aux
#define OFF_TPROB  131072
#define OFF_SID    262144
#define OFF_SPROB  294912
#define OFF_AUX    327680
// d_ws layout: Bsw hi words [0,131072) , Bsw lo words [131072,262144), expert sums f32 [262144,262208)
#define BSW_LO_WORDS 131072
#define ESUM_OFF     262144

typedef __attribute__((ext_vector_type(16))) __bf16 v16bf;
typedef __attribute__((ext_vector_type(8)))  float  v8f;

union V16 { uint32_t u[8]; v16bf v; };

__device__ __forceinline__ uint32_t bf16_rne(float f) {
  uint32_t u = __float_as_uint(f);
  return (u + 0x7FFFu + ((u >> 16) & 1u)) >> 16;
}
__device__ __forceinline__ float bf16_f32(uint32_t h) { return __uint_as_float(h << 16); }

__device__ __forceinline__ void pack_hilo(float f0, float f1, uint32_t& hi, uint32_t& lo) {
  uint32_t h0 = bf16_rne(f0), h1 = bf16_rne(f1);
  uint32_t l0 = bf16_rne(f0 - bf16_f32(h0));
  uint32_t l1 = bf16_rne(f1 - bf16_f32(h1));
  hi = (h1 << 16) | h0;
  lo = (l1 << 16) | l0;
}

__device__ __forceinline__ void pack_tile(const float4& a0, const float4& a1,
                                          const float4& a2, const float4& a3,
                                          V16& hi, V16& lo) {
  pack_hilo(a0.x, a0.y, hi.u[0], lo.u[0]);
  pack_hilo(a0.z, a0.w, hi.u[1], lo.u[1]);
  pack_hilo(a1.x, a1.y, hi.u[2], lo.u[2]);
  pack_hilo(a1.z, a1.w, hi.u[3], lo.u[3]);
  pack_hilo(a2.x, a2.y, hi.u[4], lo.u[4]);
  pack_hilo(a2.z, a2.w, hi.u[5], lo.u[5]);
  pack_hilo(a3.x, a3.y, hi.u[6], lo.u[6]);
  pack_hilo(a3.z, a3.w, hi.u[7], lo.u[7]);
}

// ---------------- W pre-swizzle into WMMA B-layout (hi/lo bf16) -------------
// B tile (kt,nt): B[k][n] = W[nt*16+n][kt*32+k].  Word index:
//   ((kt*4+nt)*32 + lane)*8 + j ; lane&15 = n, lane>>4 selects K half (+16), word j = K pair (2j,2j+1)
__global__ void moe_prep_w(const float* __restrict__ W, uint32_t* __restrict__ ws_u) {
  int t = blockIdx.x * blockDim.x + threadIdx.x;           // 131072 threads
  if (blockIdx.x == 0 && threadIdx.x < NEXP)
    ((float*)ws_u)[ESUM_OFF + threadIdx.x] = 0.0f;         // zero expert sums each launch
  int j    = t & 7;
  int lane = (t >> 3) & 31;
  int nt   = (t >> 8) & 3;
  int kt   = t >> 10;
  int expert = nt * 16 + (lane & 15);
  int k0     = kt * 32 + ((lane >> 4) * 16) + 2 * j;
  float f0 = W[(size_t)expert * DIM + k0];
  float f1 = W[(size_t)expert * DIM + k0 + 1];
  uint32_t hi, lo;
  pack_hilo(f0, f1, hi, lo);
  ws_u[t]                = hi;   // t == ((kt*4+nt)*32+lane)*8+j
  ws_u[t + BSW_LO_WORDS] = lo;
}

// ---------------- main: GEMM (split-bf16 WMMA, M=32/wave) + softmax + top-8 -
__global__ __launch_bounds__(128) void moe_router_main(const float* __restrict__ x,
                                                       const uint32_t* __restrict__ Bsw,
                                                       const float* __restrict__ gate_bias,
                                                       float* __restrict__ out,
                                                       float* __restrict__ esum_g) {
  // per wave: 32x65 logits + 32 rowmax + 32 inv_sum = 2144 floats; 4 waves + 64 expert partials
  __shared__ float lds[4 * 2144 + 64];
  float* expert_lds = lds + 4 * 2144;
  const int wave = threadIdx.x >> 5;
  const int lane = threadIdx.x & 31;
  const int half = lane >> 4;
  const int l15  = lane & 15;
  const int tokenBase = blockIdx.x * 128 + wave * 32;
  const float* xrow0 = x + (size_t)(tokenBase + l15) * DIM;        // A tile 0 rows
  const float* xrow1 = xrow0 + (size_t)16 * DIM;                   // A tile 1 rows
  const int koff = half * 16;                                      // A layout: lanes 16-31 hold K+16..K+31

  if (threadIdx.x < NEXP) expert_lds[threadIdx.x] = 0.0f;

  v8f acc[8] = {v8f{}, v8f{}, v8f{}, v8f{}, v8f{}, v8f{}, v8f{}, v8f{}};  // [tile*4 + nt]
  const uint4* bp = (const uint4*)Bsw;

  for (int kt = 0; kt < KTILES; ++kt) {
    const float4* x0p = (const float4*)(xrow0 + kt * 32 + koff);
    const float4* x0q = (const float4*)(xrow0 + kt * 32 + koff + 8);
    const float4* x1p = (const float4*)(xrow1 + kt * 32 + koff);
    const float4* x1q = (const float4*)(xrow1 + kt * 32 + koff + 8);
    float4 a0 = x0p[0], a1 = x0p[1], a2 = x0q[0], a3 = x0q[1];
    float4 b0 = x1p[0], b1 = x1p[1], b2 = x1q[0], b3 = x1q[1];
    __builtin_prefetch(xrow0 + kt * 32 + 32 + koff, 0, 0);   // speculative next-tile prefetch
    __builtin_prefetch(xrow1 + kt * 32 + 32 + koff, 0, 0);

    V16 A0hi, A0lo, A1hi, A1lo;
    pack_tile(a0, a1, a2, a3, A0hi, A0lo);
    pack_tile(b0, b1, b2, b3, A1hi, A1lo);

#pragma unroll
    for (int nt = 0; nt < 4; ++nt) {
      uint32_t base = (((uint32_t)(kt * 4 + nt) * 32u + (uint32_t)lane) * 8u) >> 2;  // uint4 index
      V16 Bhi, Blo;
      uint4 w0 = bp[base],                       w1 = bp[base + 1];
      uint4 y0 = bp[base + (BSW_LO_WORDS >> 2)], y1 = bp[base + (BSW_LO_WORDS >> 2) + 1];
      Bhi.u[0]=w0.x; Bhi.u[1]=w0.y; Bhi.u[2]=w0.z; Bhi.u[3]=w0.w;
      Bhi.u[4]=w1.x; Bhi.u[5]=w1.y; Bhi.u[6]=w1.z; Bhi.u[7]=w1.w;
      Blo.u[0]=y0.x; Blo.u[1]=y0.y; Blo.u[2]=y0.z; Blo.u[3]=y0.w;
      Blo.u[4]=y1.x; Blo.u[5]=y1.y; Blo.u[6]=y1.z; Blo.u[7]=y1.w;
      // 6 WMMAs per B load: both A tiles share B (halves L2 traffic per FLOP)
      acc[nt]     = __builtin_amdgcn_wmma_f32_16x16x32_bf16(false, A0hi.v, false, Bhi.v, (short)0, acc[nt],     false, false);
      acc[4 + nt] = __builtin_amdgcn_wmma_f32_16x16x32_bf16(false, A1hi.v, false, Bhi.v, (short)0, acc[4 + nt], false, false);
      acc[nt]     = __builtin_amdgcn_wmma_f32_16x16x32_bf16(false, A0hi.v, false, Blo.v, (short)0, acc[nt],     false, false);
      acc[4 + nt] = __builtin_amdgcn_wmma_f32_16x16x32_bf16(false, A1hi.v, false, Blo.v, (short)0, acc[4 + nt], false, false);
      acc[nt]     = __builtin_amdgcn_wmma_f32_16x16x32_bf16(false, A0lo.v, false, Bhi.v, (short)0, acc[nt],     false, false);
      acc[4 + nt] = __builtin_amdgcn_wmma_f32_16x16x32_bf16(false, A1lo.v, false, Bhi.v, (short)0, acc[4 + nt], false, false);
    }
  }

  // add bias, spill logits to LDS (row stride 65 floats: conflict-free)
  float* wreg = lds + wave * 2144;
  float* smax = wreg + 32 * 65;        // [0..31] row max
  float* sinv = smax + 32;             // [0..31] 1/sum
#pragma unroll
  for (int t = 0; t < 2; ++t) {
#pragma unroll
    for (int nt = 0; nt < 4; ++nt) {
      float gb = gate_bias[nt * 16 + l15];
#pragma unroll
      for (int r = 0; r < 8; ++r) {
        int m = t * 16 + (half ? (r + 8) : r);
        float v = acc[t * 4 + nt][r] + gb;
        acc[t * 4 + nt][r] = v;
        wreg[m * 65 + nt * 16 + l15] = v;
      }
    }
  }
  __syncthreads();

  // per-token softmax stats: all 32 lanes, one token each
  {
    const float* row = wreg + lane * 65;
    float mx = row[0];
    for (int n = 1; n < NEXP; ++n) mx = fmaxf(mx, row[n]);
    float s = 0.0f;
    for (int n = 0; n < NEXP; ++n) s += __expf(row[n] - mx);
    smax[lane] = mx;
    sinv[lane] = 1.0f / s;
  }
  __syncthreads();

  // expert mean accumulation straight from accumulator registers (lane owns expert nt*16+l15)
#pragma unroll
  for (int nt = 0; nt < 4; ++nt) {
    float ssum = 0.0f;
#pragma unroll
    for (int t = 0; t < 2; ++t) {
#pragma unroll
      for (int r = 0; r < 8; ++r) {
        int m = t * 16 + (half ? (r + 8) : r);
        ssum += __expf(acc[t * 4 + nt][r] - smax[m]) * sinv[m];
      }
    }
    atomicAdd(&expert_lds[nt * 16 + l15], ssum);
  }

  // top-8 per token (all 32 lanes); tie -> lowest index first, matching lax.top_k
  {
    int tok = tokenBase + lane;
    const float* row = wreg + lane * 65;
    float mx = smax[lane], inv = sinv[lane];
    uint64_t chosen = 0;
    for (int t = 0; t < 8; ++t) {
      float best = -3.402823466e38f; int bi = 0;
      for (int n = 0; n < NEXP; ++n) {
        if ((chosen >> n) & 1ull) continue;
        float v = row[n];
        if (v > best) { best = v; bi = n; }
      }
      chosen |= 1ull << bi;
      out[(size_t)tok * 8 + t]             = (float)bi;
      out[OFF_TPROB + (size_t)tok * 8 + t] = __expf(best - mx) * inv;
    }
    out[OFF_SID   + (size_t)tok * 2]     = 0.0f;
    out[OFF_SID   + (size_t)tok * 2 + 1] = 1.0f;
    out[OFF_SPROB + (size_t)tok * 2]     = 0.5f;
    out[OFF_SPROB + (size_t)tok * 2 + 1] = 0.5f;
  }

  __syncthreads();
  if (threadIdx.x < NEXP) atomicAdd(&esum_g[threadIdx.x], expert_lds[threadIdx.x]);
}

// ---------------- aux loss ---------------------------------------------------
__global__ void moe_aux(const float* __restrict__ esum, float* __restrict__ out) {
  __shared__ float red[64];
  float v = esum[threadIdx.x] * (1.0f / (float)NTOK);
  red[threadIdx.x] = v * v;
  __syncthreads();
  for (int s = 32; s > 0; s >>= 1) {
    if (threadIdx.x < s) red[threadIdx.x] += red[threadIdx.x + s];
    __syncthreads();
  }
  if (threadIdx.x == 0) out[OFF_AUX] = 0.01f * red[0] * (1.0f / (float)NEXP);
}

extern "C" void kernel_launch(void* const* d_in, const int* in_sizes, int n_in,
                              void* d_out, int out_size, void* d_ws, size_t ws_size,
                              hipStream_t stream) {
  const float* x  = (const float*)d_in[0];
  const float* W  = (const float*)d_in[1];
  const float* gb = (const float*)d_in[2];
  float*    out   = (float*)d_out;
  uint32_t* Bsw   = (uint32_t*)d_ws;
  float*    esum  = (float*)d_ws + ESUM_OFF;

  moe_prep_w      <<<512, 256, 0, stream>>>(W, Bsw);
  moe_router_main <<<NTOK / 128, 128, 0, stream>>>(x, Bsw, gb, out, esum);
  moe_aux         <<<1, 64, 0, stream>>>(esum, out);
}
